// GraphDetector_rnn_77979426226961
// MI455X (gfx1250) — matrix operand order repaired and checked by reading, not compiled, not run on hardware
//
#include <hip/hip_runtime.h>
#include <hip/hip_bf16.h>
#include <math.h>

// ---------------------------------------------------------------------------
// GraphDetector_rnn forward for MI455X (gfx1250, wave32, WMMA f16->f32)
// T=32, N=1024, K+1=16 neighbors, HEADS=2, conv=64, D=32, ODIM=50
// B operands pre-swizzled into WMMA fragment layout (no in-loop guards).
// ---------------------------------------------------------------------------

typedef __attribute__((ext_vector_type(16))) _Float16 v16h;
typedef __attribute__((ext_vector_type(8)))  float    v8f;

#define TT    32
#define NNODE 1024
#define KNB   16                 // neighbors incl. self loop
#define M1    (TT * NNODE)       // 32768 rows
#define M2    ((TT - 1) * NNODE) // 31744 rows
#define ODIM  50

// fast transcendentals: v_exp_f32 / v_rcp_f32, branch-free
__device__ __forceinline__ float fast_rcp(float x) {
  return __builtin_amdgcn_rcpf(x);
}
__device__ __forceinline__ float fast_tanh(float x) {
  float cx = fminf(fmaxf(x, -15.0f), 15.0f);
  float e  = __expf(2.0f * cx);
  return (e - 1.0f) * fast_rcp(e + 1.0f);
}
__device__ __forceinline__ float fast_sigmoid(float x) {
  return fast_rcp(1.0f + __expf(-x));
}

// --------------------------- neighbor recovery -----------------------------
__global__ void k_srcs(const int* __restrict__ edge, int* __restrict__ srcs) {
  int idx = blockIdx.x * blockDim.x + threadIdx.x;
  if (idx >= TT * NNODE) return;
  int t = idx >> 10;
  int n = idx & (NNODE - 1);
  const int* ep = edge + (size_t)t * NNODE * NNODE + n;
  int cnt = 0;
  for (int s = 0; s < NNODE; ++s) {
    if (ep[(size_t)s * NNODE] != 0) {
      srcs[((size_t)t * KNB + cnt) * NNODE + n] = s;
      if (++cnt == KNB - 1) break;
    }
  }
  for (; cnt < KNB; ++cnt) srcs[((size_t)t * KNB + cnt) * NNODE + n] = n;
}

// ------------------- pack f32 weight -> WMMA B fragments -------------------
// Source W: [K, Nc] row-major f32. Dest Bp element index:
//   e = (((nt*KT + kk)*32 + lane)*16 + j)
//   col = nt*16 + (lane&15), k = kk*32 + (lane>>4)*8 + (j&7) + (j>>3)*16, zero if col >= Nc.
// In the GEMM each lane then reads its 16 halfs as one contiguous v16h.
__global__ void k_pack_b(const float* __restrict__ W, _Float16* __restrict__ Bp,
                         int K, int Nc, int KT, int total) {
  int e = blockIdx.x * blockDim.x + threadIdx.x;
  if (e >= total) return;
  int j    = e & 15;
  int lane = (e >> 4) & 31;
  int kt_nt = e >> 9;            // nt*KT + kk
  int kk = kt_nt % KT;
  int nt = kt_nt / KT;
  int col = nt * 16 + (lane & 15);
  int k   = kk * 32 + (lane >> 4) * 8 + (j & 7) + (j >> 3) * 16;
  _Float16 v = (_Float16)0.0f;
  if (col < Nc && k < K) v = (_Float16)W[(size_t)k * Nc + col];
  Bp[e] = v;
}

// --------------------------- WMMA GEMM -------------------------------------
// C[M,Nc] = act( A[M,K](f32->f16) * B + bias[Nc] ), f32 accumulate.
// One 16x16 tile per wave; K fully unrolled (KVAL = 32 or 64).
// act: 0 = none, 1 = tanh
template <int KVAL>
__global__ void k_gemm(const float* __restrict__ A, const _Float16* __restrict__ Bp,
                       const float* __restrict__ bias, float* __restrict__ C,
                       int M, int Nc, int act) {
  constexpr int KT = KVAL / 32;
  const int lane = threadIdx.x & 31;
  const int wave = threadIdx.x >> 5;
  const int ntb  = (Nc + 15) >> 4;
  const int mtb  = M >> 4;
  int tile = blockIdx.x * (blockDim.x >> 5) + wave;
  if (tile >= mtb * ntb) return;            // wave-uniform: EXEC all-ones below
  const int mt   = tile / ntb;
  const int nt   = tile - mt * ntb;
  const int half = lane >> 4;
  const int l16  = lane & 15;
  const int mrow = mt * 16 + l16;
  const int ncol = nt * 16 + l16;

  const float*    Arow  = A  + (size_t)mrow * KVAL;
  const _Float16* Bbase = Bp + (((size_t)nt * KT * 32) + lane) * 16;

  v8f acc = {};
#pragma unroll
  for (int kk = 0; kk < KT; ++kk) {
    const int kt = kk * 32;
    // A tile: 16x32 f16; lane half: K = {kt+half*8+j (j<8)} U {kt+16+half*8+(j-8)}
    const float* Ap = Arow + kt + half * 8;
    v16h a;
#pragma unroll
    for (int j = 0; j < 8; ++j)  a[j] = (_Float16)Ap[j];
#pragma unroll
    for (int j = 8; j < 16; ++j) a[j] = (_Float16)Ap[j + 8];

    // B tile: one contiguous 32-byte fragment per lane (pre-swizzled, padded).
    v16h b = *(const v16h*)(Bbase + (size_t)kk * 32 * 16);

    acc = __builtin_amdgcn_wmma_f32_16x16x32_f16(
        /*neg_a=*/false, a, /*neg_b=*/false, b,
        /*c_mod=*/(short)0, acc, /*reuse_a=*/false, /*reuse_b=*/false);
  }

  if (ncol < Nc) {
    const float bv = bias ? bias[ncol] : 0.0f;
#pragma unroll
    for (int v = 0; v < 8; ++v) {
      int m = mt * 16 + v + half * 8;       // D layout: M = v + 8*(lane/16)
      float r = acc[v] + bv;
      if (act == 1) r = fast_tanh(r);
      C[(size_t)m * Nc + ncol] = r;
    }
  }
}

// --------------------------- GATv2 attention -------------------------------
template <int C>
__global__ void k_gat_attn(const float* __restrict__ gl, const float* __restrict__ gr,
                           const int* __restrict__ srcs, const float* __restrict__ att,
                           const float* __restrict__ bias, float* __restrict__ out) {
  const int HC = 2 * C;
  const int n = blockIdx.x;
  const int t = blockIdx.y;
  const int tid = threadIdx.x;              // 0 .. 2C-1  (h = tid/C, c = tid%C)

  __shared__ float s_gl[KNB][2 * C];
  __shared__ float s_gr[2 * C];
  __shared__ float s_att[2 * C];
  __shared__ float s_logit[2][KNB];
  __shared__ float s_acc[2 * C];
  __shared__ int   s_src[KNB];

  const size_t row = (size_t)t * NNODE + n;
  if (tid < KNB) s_src[tid] = srcs[((size_t)t * KNB + tid) * NNODE + n];
  s_gr[tid]  = gr[row * HC + tid];
  s_att[tid] = att[tid];
  __syncthreads();

  for (int k = 0; k < KNB; ++k) {
    int s = s_src[k];
    s_gl[k][tid] = gl[((size_t)t * NNODE + s) * HC + tid];
  }
  __syncthreads();

  if (tid < 32) {                           // one wave: 2 heads x 16 neighbors
    int k = tid & 15, h = tid >> 4;
    float l = 0.0f;
    for (int c = 0; c < C; ++c) {
      float e = s_gl[k][h * C + c] + s_gr[h * C + c];
      e = (e > 0.0f) ? e : 0.2f * e;        // leaky_relu(0.2)
      l += s_att[h * C + c] * e;
    }
    s_logit[h][k] = l;
  }
  __syncthreads();

  const int h = tid / C;
  float m = -1e30f;
#pragma unroll
  for (int k = 0; k < KNB; ++k) m = fmaxf(m, s_logit[h][k]);
  float w[KNB], denom = 0.0f;
#pragma unroll
  for (int k = 0; k < KNB; ++k) { w[k] = __expf(s_logit[h][k] - m); denom += w[k]; }
  float acc = 0.0f;
#pragma unroll
  for (int k = 0; k < KNB; ++k) acc += w[k] * s_gl[k][tid];
  s_acc[tid] = acc * fast_rcp(denom);
  __syncthreads();

  if (tid < C) {
    float r = 0.5f * (s_acc[tid] + s_acc[C + tid]) + bias[tid];
    out[row * C + tid] = (r > 0.0f) ? r : (__expf(r) - 1.0f);   // elu
  }
}

// --------------------------- fused 2-layer GRU -----------------------------
__device__ __forceinline__ void gru_step(const float* __restrict__ x, float* __restrict__ h,
                                         const float* __restrict__ Wih,
                                         const float* __restrict__ Whh,
                                         const float* __restrict__ bih,
                                         const float* __restrict__ bhh) {
  float hn[32];
  for (int d = 0; d < 32; ++d) {
    float ir = bih[d], iz = bih[32 + d], in_ = bih[64 + d];
    float hr = bhh[d], hz = bhh[32 + d], hnn = bhh[64 + d];
    for (int i = 0; i < 32; ++i) {
      float xi = x[i], hi = h[i];
      ir  += Wih[d * 32 + i] * xi;
      iz  += Wih[(32 + d) * 32 + i] * xi;
      in_ += Wih[(64 + d) * 32 + i] * xi;
      hr  += Whh[d * 32 + i] * hi;
      hz  += Whh[(32 + d) * 32 + i] * hi;
      hnn += Whh[(64 + d) * 32 + i] * hi;
    }
    float r  = fast_sigmoid(ir + hr);
    float z  = fast_sigmoid(iz + hz);
    float nv = fast_tanh(in_ + r * hnn);
    hn[d] = (1.0f - z) * nv + z * h[d];
  }
  for (int d = 0; d < 32; ++d) h[d] = hn[d];
}

__global__ void k_gru2(const float* __restrict__ xin,       // [T*N, 32]
                       const float* __restrict__ W0ih, const float* __restrict__ W0hh,
                       const float* __restrict__ b0ih, const float* __restrict__ b0hh,
                       const float* __restrict__ W1ih, const float* __restrict__ W1hh,
                       const float* __restrict__ b1ih, const float* __restrict__ b1hh,
                       float* __restrict__ E) {               // [T*N, 32]
  __shared__ float sW[4][96 * 32];
  __shared__ float sb[4][96];
  for (int i = threadIdx.x; i < 96 * 32; i += blockDim.x) {
    sW[0][i] = W0ih[i]; sW[1][i] = W0hh[i]; sW[2][i] = W1ih[i]; sW[3][i] = W1hh[i];
  }
  for (int i = threadIdx.x; i < 96; i += blockDim.x) {
    sb[0][i] = b0ih[i]; sb[1][i] = b0hh[i]; sb[2][i] = b1ih[i]; sb[3][i] = b1hh[i];
  }
  __syncthreads();
  int n = blockIdx.x * blockDim.x + threadIdx.x;
  if (n >= NNODE) return;
  float h0[32], h1[32], x[32];
  for (int i = 0; i < 32; ++i) { h0[i] = 0.0f; h1[i] = 0.0f; }
  for (int t = 0; t < TT; ++t) {
    const size_t row = (size_t)t * NNODE + n;
    for (int i = 0; i < 32; ++i) x[i] = xin[row * 32 + i];
    gru_step(x,  h0, sW[0], sW[1], sb[0], sb[1]);
    gru_step(h0, h1, sW[2], sW[3], sb[2], sb[3]);
    for (int i = 0; i < 32; ++i) E[row * 32 + i] = h1[i];
  }
}

// --------------------------- host-side launcher ----------------------------
extern "C" void kernel_launch(void* const* d_in, const int* in_sizes, int n_in,
                              void* d_out, int out_size, void* d_ws, size_t ws_size,
                              hipStream_t stream) {
  (void)in_sizes; (void)n_in; (void)out_size; (void)ws_size;

  // ---- inputs (setup_inputs order) ----
  const float* x      = (const float*)d_in[0];
  const int*   edge   = (const int*)d_in[1];
  const float* g1_Wl  = (const float*)d_in[2];
  const float* g1_bl  = (const float*)d_in[3];
  const float* g1_Wr  = (const float*)d_in[4];
  const float* g1_br  = (const float*)d_in[5];
  const float* g1_att = (const float*)d_in[6];
  const float* g1_b   = (const float*)d_in[7];
  const float* g2_Wl  = (const float*)d_in[8];
  const float* g2_bl  = (const float*)d_in[9];
  const float* g2_Wr  = (const float*)d_in[10];
  const float* g2_br  = (const float*)d_in[11];
  const float* g2_att = (const float*)d_in[12];
  const float* g2_b   = (const float*)d_in[13];
  const float* r1_Wl  = (const float*)d_in[14];
  const float* r1_bl  = (const float*)d_in[15];
  const float* r1_Wr  = (const float*)d_in[16];
  const float* r1_br  = (const float*)d_in[17];
  const float* r1_att = (const float*)d_in[18];
  const float* r1_b   = (const float*)d_in[19];
  const float* f1_Wl  = (const float*)d_in[20];
  const float* f1_bl  = (const float*)d_in[21];
  const float* f1_Wr  = (const float*)d_in[22];
  const float* f1_br  = (const float*)d_in[23];
  const float* f1_att = (const float*)d_in[24];
  const float* f1_b   = (const float*)d_in[25];
  const float* gru0_Wih = (const float*)d_in[26];
  const float* gru0_Whh = (const float*)d_in[27];
  const float* gru0_bih = (const float*)d_in[28];
  const float* gru0_bhh = (const float*)d_in[29];
  const float* gru1_Wih = (const float*)d_in[30];
  const float* gru1_Whh = (const float*)d_in[31];
  const float* gru1_bih = (const float*)d_in[32];
  const float* gru1_bhh = (const float*)d_in[33];
  const float* rec2_W = (const float*)d_in[34];
  const float* rec2_b = (const float*)d_in[35];
  const float* rec3_W = (const float*)d_in[36];
  const float* rec3_b = (const float*)d_in[37];
  const float* fore3_W = (const float*)d_in[38];
  const float* fore3_b = (const float*)d_in[39];

  // ---- outputs: recon[31*1024*50] ++ forecast[31*1024*50] ++ E[32*1024*32] ----
  float* recon    = (float*)d_out;
  float* forecast = recon + (size_t)M2 * ODIM;
  float* E        = forecast + (size_t)M2 * ODIM;

  // ---- workspace layout (4-byte units) ----
  float* wsf = (float*)d_ws;
  size_t off = 0;
  int*   srcs = (int*)d_ws;            off += (size_t)TT * KNB * NNODE;   // 2 MB
  float* gl1  = wsf + off;             off += (size_t)M1 * 128;
  float* gr1  = wsf + off;             off += (size_t)M1 * 128;
  float* act1 = wsf + off;             off += (size_t)M1 * 64;
  float* gl2  = wsf + off;             off += (size_t)M1 * 64;
  float* gr2  = wsf + off;             off += (size_t)M1 * 64;
  float* out2 = wsf + off;             off += (size_t)M1 * 32;
  _Float16* wh = (_Float16*)(wsf + off);
  // phase-2 reuse (M2*128 <= M1*128, M2*64 <= M1*64):
  float* glr  = gl1;
  float* grr  = gr1;
  float* recA = act1;
  float* recB = gl2;

  // fp16 packed-B weight slots (padded to multiple-of-16 columns)
  _Float16* h_g1Wl  = wh;  wh += 64 * 128;   // K=64, 8 ntiles
  _Float16* h_g1Wr  = wh;  wh += 64 * 128;
  _Float16* h_g2Wl  = wh;  wh += 64 * 64;    // K=64, 4 ntiles
  _Float16* h_g2Wr  = wh;  wh += 64 * 64;
  _Float16* h_r1Wl  = wh;  wh += 32 * 128;   // K=32, 8 ntiles
  _Float16* h_r1Wr  = wh;  wh += 32 * 128;
  _Float16* h_f1Wl  = wh;  wh += 32 * 128;
  _Float16* h_f1Wr  = wh;  wh += 32 * 128;
  _Float16* h_rec2  = wh;  wh += 64 * 64;    // K=64, 4 ntiles
  _Float16* h_rec3  = wh;  wh += 64 * 64;    // K=64, Nc=50 -> 4 ntiles padded
  _Float16* h_fore3 = wh;  wh += 64 * 64;

  auto pack = [&](const float* W, _Float16* Bp, int K, int Nc) {
    int ntb = (Nc + 15) / 16;
    int KT = K / 32;
    int total = ntb * KT * 512;              // ntb*KT tiles * 32 lanes * 16 halfs
    k_pack_b<<<(total + 255) / 256, 256, 0, stream>>>(W, Bp, K, Nc, KT, total);
  };
  auto gemm32 = [&](const float* A, const _Float16* Bp, const float* bias, float* C,
                    int M, int Nc, int act) {
    int tiles = (M / 16) * ((Nc + 15) / 16);
    k_gemm<32><<<(tiles + 7) / 8, 256, 0, stream>>>(A, Bp, bias, C, M, Nc, act);
  };
  auto gemm64 = [&](const float* A, const _Float16* Bp, const float* bias, float* C,
                    int M, int Nc, int act) {
    int tiles = (M / 16) * ((Nc + 15) / 16);
    k_gemm<64><<<(tiles + 7) / 8, 256, 0, stream>>>(A, Bp, bias, C, M, Nc, act);
  };

  // 0) neighbor lists + packed fp16 weights
  k_srcs<<<(TT * NNODE + 255) / 256, 256, 0, stream>>>(edge, srcs);
  pack(g1_Wl, h_g1Wl, 64, 128);  pack(g1_Wr, h_g1Wr, 64, 128);
  pack(g2_Wl, h_g2Wl, 64, 64);   pack(g2_Wr, h_g2Wr, 64, 64);
  pack(r1_Wl, h_r1Wl, 32, 128);  pack(r1_Wr, h_r1Wr, 32, 128);
  pack(f1_Wl, h_f1Wl, 32, 128);  pack(f1_Wr, h_f1Wr, 32, 128);
  pack(rec2_W, h_rec2, 64, 64);
  pack(rec3_W, h_rec3, 64, ODIM);
  pack(fore3_W, h_fore3, 64, ODIM);

  // 1) GAT layer 1: x -> act1 (elu), C=64
  gemm64(x, h_g1Wl, g1_bl, gl1, M1, 128, 0);
  gemm64(x, h_g1Wr, g1_br, gr1, M1, 128, 0);
  k_gat_attn<64><<<dim3(NNODE, TT), 128, 0, stream>>>(gl1, gr1, srcs, g1_att, g1_b, act1);

  // 2) GAT layer 2: act1 -> out2 (elu), C=32
  gemm64(act1, h_g2Wl, g2_bl, gl2, M1, 64, 0);
  gemm64(act1, h_g2Wr, g2_br, gr2, M1, 64, 0);
  k_gat_attn<32><<<dim3(NNODE, TT), 64, 0, stream>>>(gl2, gr2, srcs, g2_att, g2_b, out2);

  // 3) stacked GRUs: out2 -> E (per-node recurrence, fully parallel over nodes)
  k_gru2<<<(NNODE + 255) / 256, 256, 0, stream>>>(out2,
      gru0_Wih, gru0_Whh, gru0_bih, gru0_bhh,
      gru1_Wih, gru1_Whh, gru1_bih, gru1_bhh, E);

  // 4) recon branch: GAT r1 on E[1:], then tanh(. @ rec2) @ rec3
  gemm32(E + (size_t)NNODE * 32, h_r1Wl, r1_bl, glr, M2, 128, 0);
  gemm32(E + (size_t)NNODE * 32, h_r1Wr, r1_br, grr, M2, 128, 0);
  k_gat_attn<64><<<dim3(NNODE, TT - 1), 128, 0, stream>>>(
      glr, grr, srcs + (size_t)KNB * NNODE /* t = 1.. */, r1_att, r1_b, recA);
  gemm64(recA, h_rec2, rec2_b, recB, M2, 64, 1);            // tanh
  gemm64(recB, h_rec3, rec3_b, recon, M2, ODIM, 0);

  // 5) forecast branch: GAT f1 on E[:-1], tanh(. @ rec2) @ fore3
  gemm32(E, h_f1Wl, f1_bl, glr, M2, 128, 0);
  gemm32(E, h_f1Wr, f1_br, grr, M2, 128, 0);
  k_gat_attn<64><<<dim3(NNODE, TT - 1), 128, 0, stream>>>(
      glr, grr, srcs /* t = 0.. */, f1_att, f1_b, recA);
  gemm64(recA, h_rec2, rec2_b, recB, M2, 64, 1);            // tanh
  gemm64(recB, h_fore3, fore3_b, forecast, M2, ODIM, 0);
}
